// GeometricModalityFusion_46059229283215
// MI455X (gfx1250) — compile-verified, matrix operand order checked by reading, not run on hardware
//
#include <hip/hip_runtime.h>
#include <hip/hip_bf16.h>

// ---------------------------------------------------------------------------
// GeometricModalityFusion for MI455X (gfx1250, wave32, WMMA bf16 16x16x32)
// ---------------------------------------------------------------------------

typedef __attribute__((ext_vector_type(16))) __bf16 v16bf;
typedef __attribute__((ext_vector_type(8)))  __bf16 v8bf;
typedef __attribute__((ext_vector_type(8)))  float  v8f;

static constexpr int    B_   = 32;
static constexpr int    S_   = 1024;
static constexpr int    D_   = 512;
static constexpr int    H_   = 8;
static constexpr int    HD_  = 64;
static constexpr size_t SD_  = (size_t)S_ * D_;      // 524288 = 2^19
static constexpr size_t BSD_ = (size_t)B_ * SD_;     // 16777216
static constexpr size_t MBSD_ = 3 * BSD_;            // 50331648

__device__ __forceinline__ v16bf cat16(v8bf lo, v8bf hi) {
    return __builtin_shufflevector(lo, hi, 0,1,2,3,4,5,6,7,8,9,10,11,12,13,14,15);
}

// ---------------------------------------------------------------------------
// bf16 WMMA GEMM:  Y[R,N] = X[R,K] @ Wt[N,K]^T + bias[N]
// Block tile 128x128, 8 waves (2 M x 4 N), wave tile 64x32 (4x2 WMMA tiles).
// A fragment (16x32 bf16): lane = row (lane&15), khalf = lane>>4; two b128.
// B fragment (32x16 bf16): lane = col, same K packing (Wt is K-contiguous).
// Row remap for sequence padding: gemm row i = b*len + s maps to out row
// b*S + s; len is always a power of two here so remap is shift+mask.
// R, N multiples of 128; K multiple of 32 (true for all call sites).
// ---------------------------------------------------------------------------
__global__ __launch_bounds__(256)
void gemm_bf16_kernel(const __bf16* __restrict__ X, const __bf16* __restrict__ Wt,
                      const float* __restrict__ bias,
                      float* __restrict__ Yf, __bf16* __restrict__ Ybf,
                      int K, int N, int lenLog2)
{
    const int lane  = threadIdx.x & 31;
    const int wave  = threadIdx.x >> 5;
    const int wm    = wave & 1;
    const int wn    = wave >> 1;
    const int waveM = blockIdx.x * 128 + wm * 64;
    const int waveN = blockIdx.y * 128 + wn * 32;
    const int r16   = lane & 15;
    const int khalf = lane >> 4;
    const int lenMask = (1 << lenLog2) - 1;

    v8f acc[4][2] = {};

    const __bf16* aBase[4];
#pragma unroll
    for (int mi = 0; mi < 4; ++mi)
        aBase[mi] = X + (size_t)(waveM + mi * 16 + r16) * K + khalf * 8;
    const __bf16* bBase[2];
#pragma unroll
    for (int ni = 0; ni < 2; ++ni)
        bBase[ni] = Wt + (size_t)(waveN + ni * 16 + r16) * K + khalf * 8;

    for (int kb = 0; kb < K; kb += 32) {
        v16bf aF[4], bF[2];
#pragma unroll
        for (int mi = 0; mi < 4; ++mi) {
            v8bf lo = *(const v8bf*)(aBase[mi] + kb);
            v8bf hi = *(const v8bf*)(aBase[mi] + kb + 16);
            __builtin_prefetch(aBase[mi] + kb + 64, 0, 3);
            aF[mi] = cat16(lo, hi);
        }
#pragma unroll
        for (int ni = 0; ni < 2; ++ni) {
            v8bf lo = *(const v8bf*)(bBase[ni] + kb);
            v8bf hi = *(const v8bf*)(bBase[ni] + kb + 16);
            bF[ni] = cat16(lo, hi);
        }
#pragma unroll
        for (int mi = 0; mi < 4; ++mi)
#pragma unroll
            for (int ni = 0; ni < 2; ++ni)
                acc[mi][ni] = __builtin_amdgcn_wmma_f32_16x16x32_bf16(
                    false, aF[mi], false, bF[ni], (short)0, acc[mi][ni],
                    false, false);
    }

    // Epilogue: C/D layout -> col = lane&15, row = (lane>>4)*8 + r
    const int rowbase = khalf * 8;
#pragma unroll
    for (int mi = 0; mi < 4; ++mi) {
#pragma unroll
        for (int r = 0; r < 8; ++r) {
            const int gi = waveM + mi * 16 + rowbase + r;
            const int ob = gi >> lenLog2;          // batch index
            const int os = gi & lenMask;           // seq position
            const size_t orow = ((size_t)ob << 10) + os;   // out rows/batch = S = 1024
#pragma unroll
            for (int ni = 0; ni < 2; ++ni) {
                const int col = waveN + ni * 16 + r16;
                const float v = acc[mi][ni][r] + bias[col];
                const size_t oidx = orow * (size_t)N + col;
                if (Yf)  Yf[oidx]  = v;
                if (Ybf) Ybf[oidx] = (__bf16)v;
            }
        }
    }
}

// ---------------------------------------------------------------------------
// f32 -> bf16 conversion
// ---------------------------------------------------------------------------
__global__ void cvt_bf16_kernel(const float* __restrict__ in,
                                __bf16* __restrict__ out, size_t n)
{
    size_t i = (size_t)blockIdx.x * blockDim.x + threadIdx.x;
    size_t stride = (size_t)gridDim.x * blockDim.x;
    for (; i < n; i += stride) out[i] = (__bf16)in[i];
}

// W[K,N] f32 -> Wt[N,K] bf16
__global__ void transpose_bf16_kernel(const float* __restrict__ W,
                                      __bf16* __restrict__ Wt, int K, int N)
{
    size_t i = (size_t)blockIdx.x * blockDim.x + threadIdx.x;
    if (i >= (size_t)K * N) return;
    int k = (int)(i / N);
    int n = (int)(i - (size_t)k * N);
    Wt[(size_t)n * K + k] = (__bf16)W[i];
}

// ---------------------------------------------------------------------------
// Per-position modality attention.  One wave per (b,h,s); lane handles 2 d's.
// Flat view trick: element (h,s,d) lives at h*S*HD + s*HD + d within (S*D).
// ---------------------------------------------------------------------------
__global__ __launch_bounds__(256)
void attn_kernel(const float* __restrict__ q, const __bf16* __restrict__ kmat,
                 const __bf16* __restrict__ vmat, const float* __restrict__ tptr,
                 float* __restrict__ mha)
{
    const int lane = threadIdx.x & 31;
    const int wid  = (blockIdx.x << 3) | (threadIdx.x >> 5);
    const int s = wid & (S_ - 1);
    const int h = (wid >> 10) & (H_ - 1);
    const int b = wid >> 13;
    const int d0 = lane * 2;

    const size_t qoff = (size_t)b * SD_ + (size_t)h * (S_ * HD_) + (size_t)s * HD_ + d0;
    const float q0 = q[qoff], q1 = q[qoff + 1];

    float sc[3];
#pragma unroll
    for (int m = 0; m < 3; ++m) {
        const size_t ko = (size_t)m * BSD_ + qoff;
        float p = q0 * (float)kmat[ko] + q1 * (float)kmat[ko + 1];
#pragma unroll
        for (int off = 16; off >= 1; off >>= 1) p += __shfl_xor(p, off, 32);
        sc[m] = p;
    }
    const float t = fabsf(tptr[0]);
    const float scale = 1.0f / (8.0f * t);
    const float s0 = sc[0] * scale, s1 = sc[1] * scale, s2 = sc[2] * scale;
    const float mx = fmaxf(s0, fmaxf(s1, s2));
    const float e0 = expf(s0 - mx), e1 = expf(s1 - mx), e2 = expf(s2 - mx);
    const float inv = 1.0f / (e0 + e1 + e2);
    const float aw[3] = {e0 * inv, e1 * inv, e2 * inv};

    float o0 = 0.f, o1 = 0.f;
#pragma unroll
    for (int m = 0; m < 3; ++m) {
        const size_t vo = (size_t)m * BSD_ + qoff;
        o0 += aw[m] * (float)vmat[vo];
        o1 += aw[m] * (float)vmat[vo + 1];
    }
    const size_t mo = (size_t)b * SD_ + (size_t)s * D_ + h * HD_ + d0;
    mha[mo] = o0;
    mha[mo + 1] = o1;
}

// ---------------------------------------------------------------------------
// Angular + Cayley statistics.  Wave per (b,s): 6 Gram partials over D plus
// 3 normalized cosines; LDS reduce across 8 waves; per-block partial out
// (deterministic two-stage reduction, no float atomics).
// partials layout: part[(b*128 + blockIdx.y)*9 + j]
// ---------------------------------------------------------------------------
__global__ __launch_bounds__(256)
void stats_kernel(const float* __restrict__ feats, float* __restrict__ part)
{
    __shared__ float lds[8][9];
    const int lane = threadIdx.x & 31;
    const int wave = threadIdx.x >> 5;
    const int b = blockIdx.x;
    const int s = blockIdx.y * 8 + wave;

    const float* f0 = feats + (size_t)b * SD_ + (size_t)s * D_ + lane * 16;
    const float* f1 = f0 + BSD_;
    const float* f2 = f1 + BSD_;

    float p00 = 0, p01 = 0, p02 = 0, p11 = 0, p12 = 0, p22 = 0;
#pragma unroll
    for (int tt = 0; tt < 16; ++tt) {
        const float a = f0[tt], c = f1[tt], e = f2[tt];
        p00 += a * a; p01 += a * c; p02 += a * e;
        p11 += c * c; p12 += c * e; p22 += e * e;
    }
#pragma unroll
    for (int off = 16; off >= 1; off >>= 1) {
        p00 += __shfl_xor(p00, off, 32);
        p01 += __shfl_xor(p01, off, 32);
        p02 += __shfl_xor(p02, off, 32);
        p11 += __shfl_xor(p11, off, 32);
        p12 += __shfl_xor(p12, off, 32);
        p22 += __shfl_xor(p22, off, 32);
    }
    if (lane == 0) {
        const float n0 = sqrtf(p00), n1 = sqrtf(p11), n2 = sqrtf(p22);
        lds[wave][0] = p00; lds[wave][1] = p01; lds[wave][2] = p02;
        lds[wave][3] = p11; lds[wave][4] = p12; lds[wave][5] = p22;
        lds[wave][6] = p01 / (fmaxf(n0, 1e-12f) * fmaxf(n1, 1e-12f));
        lds[wave][7] = p02 / (fmaxf(n0, 1e-12f) * fmaxf(n2, 1e-12f));
        lds[wave][8] = p12 / (fmaxf(n1, 1e-12f) * fmaxf(n2, 1e-12f));
    }
    __syncthreads();
    if (threadIdx.x < 9) {
        float sum = 0.f;
#pragma unroll
        for (int w = 0; w < 8; ++w) sum += lds[w][threadIdx.x];
        part[((size_t)b * 128 + blockIdx.y) * 9 + threadIdx.x] = sum;
    }
}

// ---------------------------------------------------------------------------
// Per-batch fuse weights: angular (arccos/exp softmax) + closed-form Cayley
// volumes + attention_weights softmax.  coef[b*3+m] = w1*aw + w2*cw;
// coef[96] = w0.  One thread per batch.
// ---------------------------------------------------------------------------
__global__ void weights_kernel(const float* __restrict__ part,
                               const float* __restrict__ tptr,
                               const float* __restrict__ attw,
                               const float* __restrict__ rolew,
                               float* __restrict__ coef)
{
    const int b = threadIdx.x;
    // softmax(attention_weights)
    const float wa0 = attw[0], wa1 = attw[1], wa2 = attw[2];
    const float wmx = fmaxf(wa0, fmaxf(wa1, wa2));
    const float we0 = expf(wa0 - wmx), we1 = expf(wa1 - wmx), we2 = expf(wa2 - wmx);
    const float winv = 1.0f / (we0 + we1 + we2);
    const float w0 = we0 * winv, w1 = we1 * winv, w2 = we2 * winv;
    if (b == 0) coef[96] = w0;
    if (b >= B_) return;

    float st[9];
#pragma unroll
    for (int j = 0; j < 9; ++j) st[j] = 0.f;
    for (int blk = 0; blk < 128; ++blk)
#pragma unroll
        for (int j = 0; j < 9; ++j)
            st[j] += part[((size_t)b * 128 + blk) * 9 + j];

    const float t = fabsf(tptr[0]);
    const float invS = 1.0f / (float)S_;
    const float lim = 1.0f - 1e-7f;
    float cc[3] = {st[6] * invS, st[7] * invS, st[8] * invS};  // c01 c02 c12
    float ee[3];
#pragma unroll
    for (int j = 0; j < 3; ++j) {
        const float c = fminf(fmaxf(cc[j], -lim), lim);
        ee[j] = expf(-acosf(c) / t);
    }
    const float r0 = rolew[0], r1 = rolew[1], r2 = rolew[2];
    float as[3];
    as[0] = r1 * ee[0] + r2 * ee[1];
    as[1] = r0 * ee[0] + r2 * ee[2];
    as[2] = r0 * ee[1] + r1 * ee[2];
    float am = fmaxf(as[0], fmaxf(as[1], as[2]));
    float ae[3] = {expf(as[0] - am), expf(as[1] - am), expf(as[2] - am)};
    const float ainv = 1.0f / (ae[0] + ae[1] + ae[2]);

    // Cayley: Gram of flattened feats; pts: p0=f_a, p1=r(f_a+f_b), p2=f_c
    float G[3][3];
    G[0][0] = st[0]; G[0][1] = st[1]; G[0][2] = st[2];
    G[1][0] = st[1]; G[1][1] = st[3]; G[1][2] = st[4];
    G[2][0] = st[2]; G[2][1] = st[4]; G[2][2] = st[5];
    const float rr = 0.70710678118654752f;
    float vol[3];
#pragma unroll
    for (int i = 0; i < 3; ++i) {
        const int aI = i, bI = (i + 1) % 3, cI = (i + 2) % 3;
        const float Gaa = G[aI][aI], Gab = G[aI][bI], Gbb = G[bI][bI];
        const float Gac = G[aI][cI], Gbc = G[bI][cI], Gcc = G[cI][cI];
        const float q00 = Gaa;
        const float q11 = 0.5f * (Gaa + 2.f * Gab + Gbb);
        const float q22 = Gcc;
        const float q01 = rr * (Gaa + Gab);
        const float q02 = Gac;
        const float q12 = rr * (Gac + Gbc);
        const float d01 = q00 + q11 - 2.f * q01;
        const float d02 = q00 + q22 - 2.f * q02;
        const float d12 = q11 + q22 - 2.f * q12;
        vol[i] = 2.f * (d01 + d02 + d12) / 9.f;
    }
    float v0 = vol[0] / t, v1 = vol[1] / t, v2 = vol[2] / t;
    const float vm = fmaxf(v0, fmaxf(v1, v2));
    const float ce0 = expf(v0 - vm), ce1 = expf(v1 - vm), ce2 = expf(v2 - vm);
    const float cinv = 1.0f / (ce0 + ce1 + ce2);
    const float cw[3] = {ce0 * cinv, ce1 * cinv, ce2 * cinv};

#pragma unroll
    for (int m = 0; m < 3; ++m)
        coef[b * 3 + m] = w1 * (ae[m] * ainv) + w2 * cw[m];
}

// ---------------------------------------------------------------------------
// fused = w0*mha + sum_m coef[b,m]*feats[m]   ->  bf16 for the output GEMM
// ---------------------------------------------------------------------------
__global__ void fuse_kernel(const float* __restrict__ mha,
                            const float* __restrict__ feats,
                            const float* __restrict__ coef,
                            __bf16* __restrict__ outbf)
{
    const size_t i = (size_t)blockIdx.x * blockDim.x + threadIdx.x;
    if (i >= BSD_) return;
    const int b = (int)(i >> 19);   // SD_ = 2^19
    const float w0 = coef[96];
    const float v = w0 * mha[i]
                  + coef[b * 3 + 0] * feats[i]
                  + coef[b * 3 + 1] * feats[BSD_ + i]
                  + coef[b * 3 + 2] * feats[2 * BSD_ + i];
    outbf[i] = (__bf16)v;
}

// ---------------------------------------------------------------------------
extern "C" void kernel_launch(void* const* d_in, const int* in_sizes, int n_in,
                              void* d_out, int out_size, void* d_ws, size_t ws_size,
                              hipStream_t stream)
{
    (void)in_sizes; (void)n_in; (void)out_size; (void)ws_size;

    const float* text  = (const float*)d_in[0];   // (32,512,768)
    const float* image = (const float*)d_in[1];   // (32,256,1024)
    const float* audio = (const float*)d_in[2];   // (32,1024,512)
    const float* W0 = (const float*)d_in[3];  const float* b0 = (const float*)d_in[4];
    const float* W1 = (const float*)d_in[5];  const float* b1 = (const float*)d_in[6];
    const float* W2 = (const float*)d_in[7];  const float* b2 = (const float*)d_in[8];
    const float* Wq = (const float*)d_in[9];  const float* bq = (const float*)d_in[10];
    const float* Wk = (const float*)d_in[11]; const float* bk = (const float*)d_in[12];
    const float* Wv = (const float*)d_in[13]; const float* bv = (const float*)d_in[14];
    const float* Wo = (const float*)d_in[15]; const float* bo = (const float*)d_in[16];
    const float* temp  = (const float*)d_in[17];
    const float* attw  = (const float*)d_in[18];
    const float* rolew = (const float*)d_in[19];

    // ---- workspace carve-out ----
    char* ws = (char*)d_ws;
    size_t off = 0;
    auto carve = [&](size_t bytes) -> void* {
        void* p = ws + off;
        off = (off + bytes + 255) & ~(size_t)255;
        return p;
    };
    float*  featsF = (float*)carve(MBSD_ * 4);        // (M,B,S,D) f32
    float*  qf     = (float*)carve(BSD_ * 4);         // (B,S,D)   f32
    float*  mhaf   = (float*)carve(BSD_ * 4);
    float*  part   = (float*)carve((size_t)B_ * 128 * 9 * 4);
    float*  coef   = (float*)carve(4096);
    __bf16* featsB = (__bf16*)carve(MBSD_ * 2);
    __bf16* xtb    = (__bf16*)carve((size_t)32 * 512 * 768 * 2);
    __bf16* xib    = (__bf16*)carve((size_t)32 * 256 * 1024 * 2);
    __bf16* xab    = (__bf16*)carve((size_t)32 * 1024 * 512 * 2);
    __bf16* w0t    = (__bf16*)carve((size_t)768 * 512 * 2);
    __bf16* w1t    = (__bf16*)carve((size_t)1024 * 512 * 2);
    __bf16* w2t    = (__bf16*)carve((size_t)512 * 512 * 2);
    __bf16* wqt    = (__bf16*)carve((size_t)512 * 512 * 2);
    __bf16* wkt    = (__bf16*)carve((size_t)512 * 512 * 2);
    __bf16* wvt    = (__bf16*)carve((size_t)512 * 512 * 2);
    __bf16* wot    = (__bf16*)carve((size_t)512 * 512 * 2);
    __bf16* kb     = (__bf16*)carve(MBSD_ * 2);
    __bf16* vb     = (__bf16*)carve(MBSD_ * 2);
    __bf16* fusedb = (__bf16*)carve(BSD_ * 2);

    // pad rows of feats must be zero (graph-capture-safe memsets)
    hipMemsetAsync(featsF, 0, MBSD_ * 4, stream);
    hipMemsetAsync(featsB, 0, MBSD_ * 2, stream);

    // ---- bf16 conversions ----
    cvt_bf16_kernel<<<49152, 256, 0, stream>>>(text,  xtb, (size_t)32 * 512 * 768);
    cvt_bf16_kernel<<<32768, 256, 0, stream>>>(image, xib, (size_t)32 * 256 * 1024);
    cvt_bf16_kernel<<<65536, 256, 0, stream>>>(audio, xab, (size_t)32 * 1024 * 512);
    transpose_bf16_kernel<<<1536, 256, 0, stream>>>(W0, w0t, 768,  512);
    transpose_bf16_kernel<<<2048, 256, 0, stream>>>(W1, w1t, 1024, 512);
    transpose_bf16_kernel<<<1024, 256, 0, stream>>>(W2, w2t, 512,  512);
    transpose_bf16_kernel<<<1024, 256, 0, stream>>>(Wq, wqt, 512,  512);
    transpose_bf16_kernel<<<1024, 256, 0, stream>>>(Wk, wkt, 512,  512);
    transpose_bf16_kernel<<<1024, 256, 0, stream>>>(Wv, wvt, 512,  512);
    transpose_bf16_kernel<<<1024, 256, 0, stream>>>(Wo, wot, 512,  512);

    // ---- modality projections (pad via row remap; pad rows stay zero) ----
    gemm_bf16_kernel<<<dim3(128, 4), 256, 0, stream>>>(
        xtb, w0t, b0, featsF, featsB, 768, 512, 9);             // len=512
    gemm_bf16_kernel<<<dim3(64, 4), 256, 0, stream>>>(
        xib, w1t, b1, featsF + BSD_, featsB + BSD_, 1024, 512, 8);   // len=256
    gemm_bf16_kernel<<<dim3(256, 4), 256, 0, stream>>>(
        xab, w2t, b2, featsF + 2 * BSD_, featsB + 2 * BSD_, 512, 512, 10); // len=1024

    // ---- q / k / v projections (zero rows -> bias at padded positions) ----
    gemm_bf16_kernel<<<dim3(256, 4), 256, 0, stream>>>(
        featsB, wqt, bq, qf, nullptr, 512, 512, 10);
    for (int m = 0; m < 3; ++m) {
        gemm_bf16_kernel<<<dim3(256, 4), 256, 0, stream>>>(
            featsB + (size_t)m * BSD_, wkt, bk, nullptr, kb + (size_t)m * BSD_,
            512, 512, 10);
        gemm_bf16_kernel<<<dim3(256, 4), 256, 0, stream>>>(
            featsB + (size_t)m * BSD_, wvt, bv, nullptr, vb + (size_t)m * BSD_,
            512, 512, 10);
    }

    // ---- per-position modality softmax attention ----
    attn_kernel<<<32768, 256, 0, stream>>>(qf, kb, vb, temp, mhaf);

    // ---- angular + Cayley statistics, per-batch fuse weights ----
    stats_kernel<<<dim3(B_, 128), 256, 0, stream>>>(featsF, part);
    weights_kernel<<<1, 32, 0, stream>>>(part, temp, attw, rolew, coef);

    // ---- fuse branches, output projection ----
    fuse_kernel<<<65536, 256, 0, stream>>>(mhaf, featsF, coef, fusedb);
    gemm_bf16_kernel<<<dim3(256, 4), 256, 0, stream>>>(
        fusedb, wot, bo, (float*)d_out, nullptr, 512, 512, 10);
}
